// HilbertSeqWinTransBlockV5_32899449487870
// MI455X (gfx1250) — matrix operand order repaired and checked by reading, not compiled, not run on hardware
//
#include <hip/hip_runtime.h>
#include <hip/hip_bf16.h>

typedef _Float16 v16h  __attribute__((ext_vector_type(16)));
typedef _Float16 half4 __attribute__((ext_vector_type(4)));
typedef float    v8f   __attribute__((ext_vector_type(8)));

#define C_CH   64
#define TILE   128
#define HALO   16
#define RROWS  (TILE + 2 * HALO)   // 160 LDS rows
#define RS     68                  // LDS row stride in halves (136B: 34-bank stride, conflict-free)
#define BN_EPS 1e-5f

union V16H {
    v16h v;
    half4 q[4];
    _Float16 h[16];
};

// A-fragment (16x32 f16, M=seq positions, K=input channels) per CDNA5 ISA layout:
// lanes 0-15 hold M=lane, K = {0..7, 16..23}; lanes 16-31 hold M=lane-16, K = {8..15, 24..31}.
__device__ __forceinline__ v16h load_a_frag(const _Float16* __restrict__ S,
                                            int pt, int tap, int kc, int lane) {
    const int m    = lane & 15;
    const int koff = (lane >> 4) << 3;         // 0 or 8
    int rr = pt * 16 + m + tap - 1;            // tap in {0,1,2} -> shift -1..+1
    rr = rr < 0 ? 0 : (rr > RROWS - 1 ? RROWS - 1 : rr);
    const _Float16* base = S + rr * RS + kc * 32 + koff;
    V16H a;
    a.q[0] = *(const half4*)(base);            // K = koff+0..3
    a.q[1] = *(const half4*)(base + 4);        // K = koff+4..7
    a.q[2] = *(const half4*)(base + 16);       // K = koff+16..19
    a.q[3] = *(const half4*)(base + 20);       // K = koff+20..23
    return a.v;
}

// B-fragment (32x16 f16, K=input channels, N=output channels), loaded from global f32 weights.
// lanes 0-15: N=lane, K = {0..7, 16..23}; lanes 16-31: N=lane-16, K = {8..15, 24..31}.
// w layout for this block: [3 layers][64 O][64 I][3 K-taps]
__device__ __forceinline__ v16h load_b_frag(const float* __restrict__ w,
                                            int l, int ct, int tap, int kc, int lane) {
    const int co  = lane & 15;
    const int cib = kc * 32 + ((lane >> 4) << 3);
    const float* wb = w + ((size_t)((l * 64 + ct * 16 + co) * 64 + cib)) * 3 + tap;
    V16H bf;
#pragma unroll
    for (int h = 0; h < 16; ++h) bf.h[h] = (_Float16)wb[(h < 8 ? h : h + 8) * 3];
    return bf.v;
}

// One 16(pos) x 16(cout) output tile: 6 WMMAs as two independent 3-deep chains
// (split by K-chunk) for XDL pipelining; merged with co-executable VALU adds.
__device__ __forceinline__ v8f conv_tile(const _Float16* __restrict__ S,
                                         const v16h bw[3][2], int pt, int lane) {
    v16h a[3][2];
#pragma unroll
    for (int t = 0; t < 3; ++t) {
        a[t][0] = load_a_frag(S, pt, t, 0, lane);
        a[t][1] = load_a_frag(S, pt, t, 1, lane);
    }
    v8f acc0, acc1;
#pragma unroll
    for (int r = 0; r < 8; ++r) { acc0[r] = 0.f; acc1[r] = 0.f; }
#pragma unroll
    for (int t = 0; t < 3; ++t) {
        acc0 = __builtin_amdgcn_wmma_f32_16x16x32_f16(
            false, a[t][0], false, bw[t][0], (short)0, acc0, false, false);
        acc1 = __builtin_amdgcn_wmma_f32_16x16x32_f16(
            false, a[t][1], false, bw[t][1], (short)0, acc1, false, false);
    }
#pragma unroll
    for (int r = 0; r < 8; ++r) acc0[r] += acc1[r];
    return acc0;
}

__global__ __launch_bounds__(256)
void hilbert_mcconv_block_kernel(const float* __restrict__ x,
                                 const int*   __restrict__ idx_pa,
                                 const float* __restrict__ w,     // [3,64,64,3] for this block
                                 const float* __restrict__ bn_g,  // [3,64]
                                 const float* __restrict__ bn_b,
                                 const float* __restrict__ bn_m,
                                 const float* __restrict__ bn_v,
                                 float* __restrict__ out,
                                 int Npts) {
    __shared__ _Float16 sX[RROWS * RS];  // gathered input (residual source), f16
    __shared__ _Float16 sA[RROWS * RS];  // layer-1 output
    __shared__ _Float16 sB[RROWS * RS];  // layer-2 output

    const int tid  = threadIdx.x;
    const int lane = tid & 31;
    const int wave = tid >> 5;
    const int b    = blockIdx.y;
    const int tile_start = blockIdx.x * TILE;
    const size_t bofs = (size_t)b * (size_t)Npts;

    // ---- Stage 1: gather x[pa[pos]] for rows [-HALO, TILE+HALO), cvt f32->f16 into sX ----
    for (int rr = tid >> 3; rr < RROWS; rr += 32) {
        const int c8  = (tid & 7) * 8;
        const int pos = tile_start - HALO + rr;
        union { half4 q; _Float16 e[4]; } h0, h1;
#pragma unroll
        for (int i = 0; i < 4; ++i) { h0.e[i] = (_Float16)0.f; h1.e[i] = (_Float16)0.f; }
        if (pos >= 0 && pos < Npts) {
            const int g = idx_pa[bofs + pos];
            const float* src = x + (bofs + (size_t)g) * C_CH + c8;
            const float4 f0 = *(const float4*)(src);
            const float4 f1 = *(const float4*)(src + 4);
            h0.e[0] = (_Float16)f0.x; h0.e[1] = (_Float16)f0.y;
            h0.e[2] = (_Float16)f0.z; h0.e[3] = (_Float16)f0.w;
            h1.e[0] = (_Float16)f1.x; h1.e[1] = (_Float16)f1.y;
            h1.e[2] = (_Float16)f1.z; h1.e[3] = (_Float16)f1.w;
        }
        *(half4*)(sX + rr * RS + c8)     = h0.q;
        *(half4*)(sX + rr * RS + c8 + 4) = h1.q;
    }

    // ---- Per-lane folded BN params (each wave owns a fixed 16-channel column group) ----
    const int ct  = wave & 3;                  // output-channel tile 0..3
    const int col = ct * 16 + (lane & 15);     // this lane's output channel
    float scl[3], sft[3];
#pragma unroll
    for (int l = 0; l < 3; ++l) {
        const float g  = bn_g[l * 64 + col];
        const float vv = bn_v[l * 64 + col];
        const float s  = g * rsqrtf(vv + BN_EPS);
        scl[l] = s;
        sft[l] = bn_b[l * 64 + col] - bn_m[l * 64 + col] * s;
    }

    __syncthreads();

    const _Float16* srcs[2] = { sX, sA };
    _Float16*       dsts[2] = { sA, sB };
    const int mofs = (lane >> 4) << 3;

    // ---- Layers 0,1: conv+BN+ReLU over all 10 position tiles (incl. halo) ----
#pragma unroll
    for (int l = 0; l < 2; ++l) {
        v16h bw[3][2];
#pragma unroll
        for (int t = 0; t < 3; ++t)
#pragma unroll
            for (int k = 0; k < 2; ++k)
                bw[t][k] = load_b_frag(w, l, ct, t, k, lane);

        const _Float16* S = srcs[l];
        _Float16*       D = dsts[l];
        const int pt0 = wave >> 2;             // even or odd position tiles
#pragma unroll
        for (int j = 0; j < 5; ++j) {
            const int pt = pt0 + 2 * j;        // 0..9
            const v8f acc = conv_tile(S, bw, pt, lane);
            // epilogue: BN + ReLU -> f16 LDS
#pragma unroll
            for (int r = 0; r < 8; ++r) {
                float vv = acc[r] * scl[l] + sft[l];
                vv = fmaxf(vv, 0.f);
                D[(pt * 16 + r + mofs) * RS + col] = (_Float16)vv;
            }
        }
        __syncthreads();
    }

    // ---- Layer 2: conv+BN, residual add, ReLU, scatter-store to global via pa ----
    {
        v16h bw[3][2];
#pragma unroll
        for (int t = 0; t < 3; ++t)
#pragma unroll
            for (int k = 0; k < 2; ++k)
                bw[t][k] = load_b_frag(w, 2, ct, t, k, lane);

        const int pt0 = 1 + (wave >> 2);       // only the 8 interior tiles (rows 16..143)
#pragma unroll
        for (int j = 0; j < 4; ++j) {
            const int pt = pt0 + 2 * j;        // 1..8
            const v8f acc = conv_tile(sB, bw, pt, lane);
#pragma unroll
            for (int r = 0; r < 8; ++r) {
                const int row = pt * 16 + r + mofs;
                float vv = acc[r] * scl[2] + sft[2];
                const float xres = (float)sX[row * RS + col];
                vv = fmaxf(xres + vv, 0.f);
                const int pos = tile_start - HALO + row;      // guaranteed in [0, Npts)
                const int gp  = idx_pa[bofs + pos];           // scatter with pa (== inverse of re)
                out[(bofs + (size_t)gp) * C_CH + col] = vv;
            }
        }
    }
}

extern "C" void kernel_launch(void* const* d_in, const int* in_sizes, int n_in,
                              void* d_out, int out_size, void* d_ws, size_t ws_size,
                              hipStream_t stream) {
    // setup_inputs order: x, idx_pa_1, idx_re_1, idx_pa_2, idx_re_2,
    //                     conv_w, bn_gamma, bn_beta, bn_mean, bn_var
    const float* x      = (const float*)d_in[0];
    const int*   pa1    = (const int*)d_in[1];
    const int*   pa2    = (const int*)d_in[3];
    const float* conv_w = (const float*)d_in[5];
    const float* bn_g   = (const float*)d_in[6];
    const float* bn_b   = (const float*)d_in[7];
    const float* bn_m   = (const float*)d_in[8];
    const float* bn_v   = (const float*)d_in[9];
    float* out = (float*)d_out;
    float* tmp = (float*)d_ws;   // intermediate x between blocks (B*N*C f32)

    const int Bsz = 2;
    const int N   = 262144;
    dim3 grid(N / TILE, Bsz), block(256);

    // Block 0: x -> tmp
    hilbert_mcconv_block_kernel<<<grid, block, 0, stream>>>(
        x, pa1, conv_w, bn_g, bn_b, bn_m, bn_v, tmp, N);

    // Block 1: tmp -> out (second block's weights / BN params)
    const size_t wofs  = (size_t)3 * 64 * 64 * 3;
    const size_t bnofs = (size_t)3 * 64;
    hilbert_mcconv_block_kernel<<<grid, block, 0, stream>>>(
        tmp, pa2, conv_w + wofs, bn_g + bnofs, bn_b + bnofs,
        bn_m + bnofs, bn_v + bnofs, out, N);
}
